// multiHeadAttention_44272522887893
// MI455X (gfx1250) — matrix operand order repaired and checked by reading, not compiled, not run on hardware
//
#include <hip/hip_runtime.h>
#include <stdint.h>

// ---------------------------------------------------------------------------
// Problem constants
// ---------------------------------------------------------------------------
static constexpr int S_LEN = 577;
static constexpr int BATCH = 32;
static constexpr int EMB   = 1024;
static constexpr int NH    = 16;
static constexpr int HD    = 64;            // head dim
static constexpr int SP    = 608;           // S padded to multiple of 32
static constexpr int MT    = SP / 16;       // 38 q-row tiles
static constexpr int TT    = SP / 32;       // 19 kv tiles
static constexpr int BS    = BATCH * S_LEN; // 18464 (multiple of 16)
static constexpr int NQKV  = 3 * NH * HD;   // 3072

// ---------------------------------------------------------------------------
// WMMA types / helpers
// ---------------------------------------------------------------------------
typedef __attribute__((ext_vector_type(16))) __bf16 v16bf;
typedef __attribute__((ext_vector_type(8)))  float  v8f;

union FragBF {
    uint4 u[2];
    v16bf v;
};

__device__ __forceinline__ unsigned short f2bf(float f) {
    union { float f; uint32_t u; } x; x.f = f;
    uint32_t r = x.u + 0x7FFFu + ((x.u >> 16) & 1u);   // round to nearest even
    return (unsigned short)(r >> 16);
}

// A fragment (16x32 bf16, row-major source): two 16B chunks at k + c0, k + c0 + 16
//   c0 = (lane>=16) ? 8 : 0 ; row = lane & 15
// B fragment (32x16 bf16, column n = source row, K contiguous): two 16B chunks
//   at k + b0, k + b0 + 16, b0 = (lane>=16) ? 8 : 0
__device__ __forceinline__ void load_a(FragBF& f, const unsigned short* p) {
    f.u[0] = *(const uint4*)(p);
    f.u[1] = *(const uint4*)(p + 16);
}
__device__ __forceinline__ void load_b(FragBF& f, const unsigned short* p) {
    f.u[0] = *(const uint4*)(p);
    f.u[1] = *(const uint4*)(p + 16);
}

// CDNA5 async global->LDS copy, 16 bytes per lane. Tracked by ASYNCcnt.
// LDS byte address comes from the vdst VGPR; global addr = VGPR pair (GV mode, SADDR=off).
__device__ __forceinline__ void async_copy_b128(uint32_t lds_byte_off, const void* gptr) {
    asm volatile("global_load_async_to_lds_b128 %0, %1, off"
                 :: "v"(lds_byte_off), "v"(gptr)
                 : "memory");
}
__device__ __forceinline__ void wait_async0() {
    asm volatile("s_wait_asynccnt 0x0" ::: "memory");
}

// Software-pipelined wave-level GEMM: 16(M) x 64(N) f32 tile, K = EMB,
// A row-major (lda=EMB, ap pre-offset by +c0), B column-major rows
// (ldb=EMB, bp pre-offset by +b0). Double-buffered fragments; sched_barrier
// fences pin the pipeline so next-step loads issue BEFORE the current WMMA
// group (otherwise the scheduler sinks every load to just before its WMMA
// and serializes on s_wait_loadcnt 0x0).
__device__ __forceinline__ void wave_gemm_16x64(const unsigned short* __restrict__ ap,
                                                const unsigned short* __restrict__ bp,
                                                v8f acc[4]) {
    FragBF aA, aB, bA[4], bB[4];
    load_a(aA, ap);
#pragma unroll
    for (int j = 0; j < 4; ++j) load_b(bA[j], bp + (size_t)j * 16 * EMB);

    for (int k = 0; k < EMB; k += 64) {
        const int k1 = k + 32;
        load_a(aB, ap + k1);
#pragma unroll
        for (int j = 0; j < 4; ++j) load_b(bB[j], bp + (size_t)j * 16 * EMB + k1);
        __builtin_amdgcn_sched_barrier(0);   // loads(k+32) stay above WMMAs(k)
#pragma unroll
        for (int j = 0; j < 4; ++j)
            acc[j] = __builtin_amdgcn_wmma_f32_16x16x32_bf16(
                false, aA.v, false, bA[j].v, (short)0, acc[j], false, false);
        __builtin_amdgcn_sched_barrier(0);

        const int k2 = (k + 64 < EMB) ? k + 64 : 0;   // last prefetch is dummy
        load_a(aA, ap + k2);
#pragma unroll
        for (int j = 0; j < 4; ++j) load_b(bA[j], bp + (size_t)j * 16 * EMB + k2);
        __builtin_amdgcn_sched_barrier(0);   // loads(k+64) stay above WMMAs(k+32)
#pragma unroll
        for (int j = 0; j < 4; ++j)
            acc[j] = __builtin_amdgcn_wmma_f32_16x16x32_bf16(
                false, aB.v, false, bB[j].v, (short)0, acc[j], false, false);
        __builtin_amdgcn_sched_barrier(0);
    }
}

// ---------------------------------------------------------------------------
// Workspace layout (bytes). concat aliases xb (xb dead after QKV GEMM).
// ---------------------------------------------------------------------------
static constexpr size_t XB_ELEMS  = (size_t)BATCH * S_LEN * EMB;  // 18,907,136
static constexpr size_t WT_ELEMS  = (size_t)NQKV * EMB;           //  3,145,728
static constexpr size_t WOB_ELEMS = (size_t)EMB * EMB;
static constexpr size_t QKV_ELEMS = (size_t)BATCH * NH * SP * HD; // 19,922,944

static constexpr size_t OFF_XB  = 0;
static constexpr size_t OFF_WT  = OFF_XB  + XB_ELEMS  * 2;
static constexpr size_t OFF_WOB = OFF_WT  + WT_ELEMS  * 2;
static constexpr size_t OFF_Q   = OFF_WOB + WOB_ELEMS * 2;
static constexpr size_t OFF_K   = OFF_Q   + QKV_ELEMS * 2;
static constexpr size_t OFF_V   = OFF_K   + QKV_ELEMS * 2;
static constexpr size_t OFF_CC  = OFF_XB; // alias

// ---------------------------------------------------------------------------
// Stage 0: f32 -> bf16 conversions / weight repacking
// ---------------------------------------------------------------------------
__global__ void mha_cvt_x(const float* __restrict__ x, unsigned short* __restrict__ xb) {
    size_t i = (size_t)blockIdx.x * blockDim.x + threadIdx.x;
    if (i < XB_ELEMS) xb[i] = f2bf(x[i]);
}

// Wt[n, e] with n = (which*NH + h)*HD + d  <-  W_which[h, e, d]
__global__ void mha_cvt_wqkv(const float* __restrict__ Wq, const float* __restrict__ Wk,
                             const float* __restrict__ Wv, unsigned short* __restrict__ Wt) {
    size_t i = (size_t)blockIdx.x * blockDim.x + threadIdx.x;
    if (i >= WT_ELEMS) return;
    int n = (int)(i / EMB);
    int e = (int)(i % EMB);
    int which = n / (NH * HD);
    int h     = (n / HD) % NH;
    int d     = n % HD;
    const float* W = (which == 0) ? Wq : (which == 1) ? Wk : Wv;
    Wt[i] = f2bf(W[((size_t)h * EMB + e) * HD + d]);
}

__global__ void mha_cvt_wo(const float* __restrict__ Wo, unsigned short* __restrict__ Wob) {
    size_t i = (size_t)blockIdx.x * blockDim.x + threadIdx.x;
    if (i < WOB_ELEMS) Wob[i] = f2bf(Wo[i]);
}

// ---------------------------------------------------------------------------
// Stage 1: QKV projection GEMM  C[B*Sp, 3072] = xb[B*S, E] x Wt^T
// One wave computes a 16x64 tile. Epilogue scatters into Q (scaled), K, V^T.
// ---------------------------------------------------------------------------
__global__ void __launch_bounds__(256) mha_qkv(
        const unsigned short* __restrict__ xb, const unsigned short* __restrict__ Wt,
        unsigned short* __restrict__ Qb, unsigned short* __restrict__ Kb,
        unsigned short* __restrict__ Vt) {
    int wave = (int)((blockIdx.x * blockDim.x + threadIdx.x) >> 5);
    int lane = threadIdx.x & 31;
    int nt  = wave % (NQKV / 64);       // 0..47
    int tmp = wave / (NQKV / 64);
    int mt  = tmp % MT;                 // 0..37
    int b   = tmp / MT;
    if (b >= BATCH) return;             // wave-uniform

    int r  = lane & 15;
    int c0 = (lane & 16) ? 8 : 0;
    int b0 = (lane & 16) ? 8 : 0;

    int srow = mt * 16 + r;
    int arow = b * S_LEN + (srow < S_LEN ? srow : S_LEN - 1);  // clamp pad rows
    const unsigned short* ap = xb + (size_t)arow * EMB + c0;
    const unsigned short* bp = Wt + (size_t)(nt * 64 + r) * EMB + b0;

    v8f acc[4] = {};
    wave_gemm_16x64(ap, bp, acc);

    int nbase = nt * 64;
    int which = nbase / (NH * HD);
    int h     = (nbase / HD) % NH;
    size_t hb = (size_t)(b * NH + h);
#pragma unroll
    for (int j = 0; j < 4; ++j) {
#pragma unroll
        for (int i = 0; i < 8; ++i) {
            int M = (lane < 16) ? i : i + 8;
            int s = mt * 16 + M;
            int d = j * 16 + r;
            float v = acc[j][i];
            if (which == 0)       Qb[(hb * SP + s) * HD + d] = f2bf(v * 0.125f); // *D^-0.5
            else if (which == 1)  Kb[(hb * SP + s) * HD + d] = f2bf(v);
            else                  Vt[(hb * HD + d) * SP + s] = f2bf(v);
        }
    }
}

// ---------------------------------------------------------------------------
// Stage 2: flash attention. Two waves per block share one (b,h); each wave
// owns a 16-row Q tile. K and V tiles are staged cooperatively into LDS with
// CDNA5 async global->LDS copies (ASYNCcnt), then consumed as WMMA fragments
// via ds_load. Online softmax; P restaged via per-wave LDS region.
// ---------------------------------------------------------------------------
__global__ void __launch_bounds__(64) mha_attn(
        const unsigned short* __restrict__ Qb, const unsigned short* __restrict__ Kb,
        const unsigned short* __restrict__ Vt, unsigned short* __restrict__ Cc) {
    __shared__ unsigned short sK[32 * 64];      // K tile   [t_local][d]   4KB
    __shared__ unsigned short sV[64 * 32];      // V^T tile [d][t_local]   4KB
    __shared__ unsigned short sP[2 * 16 * 32];  // per-wave P staging      2KB

    int tid  = threadIdx.x;          // 0..63
    int wid  = tid >> 5;
    int lane = tid & 31;
    int qt = blockIdx.x * 2 + wid;   // 0..37 (grid.x = 19)
    int bh = blockIdx.y;             // 0..511
    unsigned short* pl = sP + wid * 16 * 32;

    const unsigned short* Qp = Qb + (size_t)bh * SP * HD;
    const unsigned short* Kp = Kb + (size_t)bh * SP * HD;
    const unsigned short* Vp = Vt + (size_t)bh * HD * SP;

    int r  = lane & 15;
    int c0 = (lane & 16) ? 8 : 0;
    int b0 = (lane & 16) ? 8 : 0;

    FragBF qf0, qf1;   // Q tile 16x64 -> two A fragments (d 0..31, 32..63)
    {
        const unsigned short* p = Qp + (size_t)(qt * 16 + r) * HD + c0;
        load_a(qf0, p);
        load_a(qf1, p + 32);
    }

    v8f o[4] = {};
    float rm[8], rl[8];
#pragma unroll
    for (int i = 0; i < 8; ++i) { rm[i] = -3.0e38f; rl[i] = 0.0f; }

    for (int tt = 0; tt < TT; ++tt) {
        // make sure the other wave finished reading the previous K/V tiles
        __syncthreads();

        // ---- async-stage K (32x64) and V^T (64x32) tiles into LDS ----
        // K tile: 256 x 16B chunks; V tile: 256 x 16B chunks; 64 threads.
#pragma unroll
        for (int it = 0; it < 4; ++it) {
            int i   = tid + it * 64;                 // 0..255
            int row = i >> 3, ch = i & 7;            // row in [0,32), chunk in [0,8)
            const unsigned short* g = Kp + (size_t)(tt * 32 + row) * HD + ch * 8;
            async_copy_b128((uint32_t)(uintptr_t)(sK + row * 64 + ch * 8), g);
        }
#pragma unroll
        for (int it = 0; it < 4; ++it) {
            int i = tid + it * 64;                   // 0..255
            int d = i >> 2, ch = i & 3;              // d in [0,64), chunk in [0,4)
            const unsigned short* g = Vp + (size_t)d * SP + tt * 32 + ch * 8;
            async_copy_b128((uint32_t)(uintptr_t)(sV + d * 32 + ch * 8), g);
        }
        wait_async0();
        __syncthreads();

        // ---- read WMMA fragments from LDS (keep as one ds clause group) ----
        FragBF kf[2][2], vf[4];
#pragma unroll
        for (int j = 0; j < 2; ++j) {
            const unsigned short* kp = sK + (j * 16 + r) * 64 + b0;
            load_b(kf[j][0], kp);
            load_b(kf[j][1], kp + 32);
        }
#pragma unroll
        for (int k = 0; k < 4; ++k)
            load_b(vf[k], sV + (k * 16 + r) * 32 + b0);
        __builtin_amdgcn_sched_barrier(0);

        // ---- scores: S = Q (16x64) x K^T (64x32), two chained K=32 WMMAs ----
        v8f sc[2];
#pragma unroll
        for (int j = 0; j < 2; ++j) {
            v8f z = {};
            z = __builtin_amdgcn_wmma_f32_16x16x32_bf16(false, qf0.v, false, kf[j][0].v, (short)0, z, false, false);
            z = __builtin_amdgcn_wmma_f32_16x16x32_bf16(false, qf1.v, false, kf[j][1].v, (short)0, z, false, false);
            int t = tt * 32 + j * 16 + r;      // column index = kv position
            if (t >= S_LEN) {
#pragma unroll
                for (int i = 0; i < 8; ++i) z[i] = -3.0e38f;
            }
            sc[j] = z;
        }

        // ---- online softmax: row reductions within each 16-lane half ----
        float nscale[8];
#pragma unroll
        for (int i = 0; i < 8; ++i) {
            float lm = fmaxf(sc[0][i], sc[1][i]);
#pragma unroll
            for (int m = 8; m >= 1; m >>= 1) lm = fmaxf(lm, __shfl_xor(lm, m, 32));
            float mnew = fmaxf(rm[i], lm);
            float p0 = __expf(sc[0][i] - mnew);
            float p1 = __expf(sc[1][i] - mnew);
            float ls = p0 + p1;
#pragma unroll
            for (int m = 8; m >= 1; m >>= 1) ls += __shfl_xor(ls, m, 32);
            float so = __expf(rm[i] - mnew);
            rl[i] = rl[i] * so + ls;
            rm[i] = mnew;
            nscale[i] = so;
            sc[0][i] = p0;
            sc[1][i] = p1;
        }
#pragma unroll
        for (int k = 0; k < 4; ++k)
#pragma unroll
            for (int i = 0; i < 8; ++i) o[k][i] *= nscale[i];

        // ---- restage P (C layout) to per-wave LDS as row-major 16x32 bf16 ----
#pragma unroll
        for (int j = 0; j < 2; ++j)
#pragma unroll
            for (int i = 0; i < 8; ++i) {
                int M = (lane < 16) ? i : i + 8;
                pl[M * 32 + j * 16 + r] = f2bf(sc[j][i]);
            }
        asm volatile("s_wait_dscnt 0x0" ::: "memory");

        FragBF pf;   // A fragment of P (16x32, K = t)
        pf.u[0] = *(const uint4*)(pl + r * 32 + c0);
        pf.u[1] = *(const uint4*)(pl + r * 32 + c0 + 16);

        // ---- O += P x V ----
#pragma unroll
        for (int k = 0; k < 4; ++k)
            o[k] = __builtin_amdgcn_wmma_f32_16x16x32_bf16(
                false, pf.v, false, vf[k].v, (short)0, o[k], false, false);
    }

    // normalize + scatter to concat buffer [B*S, E], head h at cols h*64..h*64+63
    int b = bh / NH, h = bh % NH;
#pragma unroll
    for (int k = 0; k < 4; ++k)
#pragma unroll
        for (int i = 0; i < 8; ++i) {
            int M = (lane < 16) ? i : i + 8;
            int s = qt * 16 + M;
            if (s < S_LEN) {
                float v = o[k][i] / rl[i];
                Cc[((size_t)(b * S_LEN + s)) * EMB + h * HD + k * 16 + r] = f2bf(v);
            }
        }
}

// ---------------------------------------------------------------------------
// Stage 3: output projection  out[B*S, E] = Cc x Wo^T + bo   (f32 out)
// ---------------------------------------------------------------------------
__global__ void __launch_bounds__(256) mha_proj(
        const unsigned short* __restrict__ Cc, const unsigned short* __restrict__ Wob,
        const float* __restrict__ bo, float* __restrict__ out) {
    int wave = (int)((blockIdx.x * blockDim.x + threadIdx.x) >> 5);
    int lane = threadIdx.x & 31;
    int nt = wave % (EMB / 64);
    int mt = wave / (EMB / 64);
    if (mt >= BS / 16) return;   // wave-uniform

    int r  = lane & 15;
    int c0 = (lane & 16) ? 8 : 0;
    int b0 = (lane & 16) ? 8 : 0;

    const unsigned short* ap = Cc  + (size_t)(mt * 16 + r) * EMB + c0;
    const unsigned short* bp = Wob + (size_t)(nt * 64 + r) * EMB + b0;

    v8f acc[4] = {};
    wave_gemm_16x64(ap, bp, acc);

#pragma unroll
    for (int j = 0; j < 4; ++j) {
        int n = nt * 64 + j * 16 + r;
        float bias = bo[n];
#pragma unroll
        for (int i = 0; i < 8; ++i) {
            int M = (lane < 16) ? i : i + 8;
            out[(size_t)(mt * 16 + M) * EMB + n] = acc[j][i] + bias;
        }
    }
}

// ---------------------------------------------------------------------------
// Host launcher
// ---------------------------------------------------------------------------
extern "C" void kernel_launch(void* const* d_in, const int* in_sizes, int n_in,
                              void* d_out, int out_size, void* d_ws, size_t ws_size,
                              hipStream_t stream) {
    (void)in_sizes; (void)n_in; (void)out_size; (void)ws_size;
    const float* x  = (const float*)d_in[0];
    const float* Wq = (const float*)d_in[1];
    const float* Wk = (const float*)d_in[2];
    const float* Wv = (const float*)d_in[3];
    const float* Wo = (const float*)d_in[4];
    const float* bo = (const float*)d_in[5];
    float* out = (float*)d_out;

    char* ws = (char*)d_ws;
    unsigned short* xb  = (unsigned short*)(ws + OFF_XB);
    unsigned short* Wt  = (unsigned short*)(ws + OFF_WT);
    unsigned short* Wob = (unsigned short*)(ws + OFF_WOB);
    unsigned short* Qb  = (unsigned short*)(ws + OFF_Q);
    unsigned short* Kb  = (unsigned short*)(ws + OFF_K);
    unsigned short* Vt  = (unsigned short*)(ws + OFF_V);
    unsigned short* Cc  = (unsigned short*)(ws + OFF_CC);  // aliases xb (safe: xb dead)

    // Stage 0: conversions
    mha_cvt_x   <<<(unsigned)((XB_ELEMS  + 255) / 256), 256, 0, stream>>>(x, xb);
    mha_cvt_wqkv<<<(unsigned)((WT_ELEMS  + 255) / 256), 256, 0, stream>>>(Wq, Wk, Wv, Wt);
    mha_cvt_wo  <<<(unsigned)((WOB_ELEMS + 255) / 256), 256, 0, stream>>>(Wo, Wob);

    // Stage 1: QKV GEMM — BATCH*MT*(NQKV/64) waves, 8 waves per block
    {
        unsigned waves = BATCH * MT * (NQKV / 64);  // 58368
        mha_qkv<<<waves / 8, 256, 0, stream>>>(xb, Wt, Qb, Kb, Vt);
    }

    // Stage 2: flash attention — grid (MT/2, B*H), 2 waves per block
    mha_attn<<<dim3(MT / 2, BATCH * NH), 64, 0, stream>>>(Qb, Kb, Vt, Cc);

    // Stage 3: output projection — (BS/16)*(EMB/64) waves, 8 per block
    {
        unsigned waves = (BS / 16) * (EMB / 64);    // 18464
        mha_proj<<<waves / 8, 256, 0, stream>>>(Cc, Wob, bo, out);
    }
}
